// VectorQuantizer_58531814309955
// MI455X (gfx1250) — matrix operand order repaired and checked by reading, not compiled, not run on hardware
//
#include <hip/hip_runtime.h>

#define TOKENS 32768
#define DIM 256
#define KCODES 8192

typedef __attribute__((ext_vector_type(16))) __bf16 v16bf;
typedef __attribute__((ext_vector_type(8)))  float  v8f;

union BfTile {
    v16bf v;
    unsigned int u[8];
};

// round-to-nearest-even f32 -> bf16 (bits), plus residual ("lo") term
__device__ inline void split_bf16(float f, unsigned short& hi, unsigned short& lo) {
    unsigned int u = __float_as_uint(f);
    unsigned int r = (u + 0x7FFFu + ((u >> 16) & 1u)) >> 16;
    hi = (unsigned short)r;
    float fh = __uint_as_float(r << 16);
    float rem = f - fh;
    unsigned int u2 = __float_as_uint(rem);
    lo = (unsigned short)((u2 + 0x7FFFu + ((u2 >> 16) & 1u)) >> 16);
}

// ---------------------------------------------------------------------------
// Kernel 1: prepack codebook f32 -> bf16 hi/lo arrays [K][D], and cnorm[K]
// (cnorm computed from exact f32 values). One wave per codebook row.
// ---------------------------------------------------------------------------
__global__ void __launch_bounds__(256) vq_pack_cnorm(
    const float* __restrict__ cb,
    unsigned short* __restrict__ cb_hi,
    unsigned short* __restrict__ cb_lo,
    float* __restrict__ cnorm)
{
    const int wave = threadIdx.x >> 5;
    const int lane = threadIdx.x & 31;
    const int row  = blockIdx.x * 8 + wave;

    const float* r = cb + row * DIM + lane * 8;
    float4 f0 = *(const float4*)(r);
    float4 f1 = *(const float4*)(r + 4);
    float v[8] = {f0.x, f0.y, f0.z, f0.w, f1.x, f1.y, f1.z, f1.w};

    unsigned int uh[4], ul[4];
    float s = 0.f;
#pragma unroll
    for (int i = 0; i < 4; ++i) {
        unsigned short h0, l0, h1, l1;
        split_bf16(v[2 * i + 0], h0, l0);
        split_bf16(v[2 * i + 1], h1, l1);
        uh[i] = (unsigned int)h0 | ((unsigned int)h1 << 16);
        ul[i] = (unsigned int)l0 | ((unsigned int)l1 << 16);
        s += v[2 * i] * v[2 * i] + v[2 * i + 1] * v[2 * i + 1];
    }
    *(uint4*)(cb_hi + row * DIM + lane * 8) = make_uint4(uh[0], uh[1], uh[2], uh[3]);
    *(uint4*)(cb_lo + row * DIM + lane * 8) = make_uint4(ul[0], ul[1], ul[2], ul[3]);

#pragma unroll
    for (int off = 16; off > 0; off >>= 1)
        s += __shfl_xor(s, off, 32);
    if (lane == 0) cnorm[row] = s;
}

// ---------------------------------------------------------------------------
// Kernel 2: fused distance + argmin with 2-way M register blocking.
// Each wave owns 32 token rows as TWO resident 16x256 A tiles (bf16 hi/lo),
// streams all 8192 codes in 16-wide tiles, and applies each loaded B tile to
// both A tiles: 48 WMMAs per 33 VMEM loads, two independent accumulator
// chains. dist = cnorm - 2*S, running per-lane (min, argmin), shfl-reduce.
// ---------------------------------------------------------------------------
__global__ void __launch_bounds__(256) vq_argmin(
    const float* __restrict__ x,
    const unsigned short* __restrict__ cb_hi,
    const unsigned short* __restrict__ cb_lo,
    const float* __restrict__ cnorm,
    int* __restrict__ out_idx)
{
    const int wave  = threadIdx.x >> 5;
    const int lane  = threadIdx.x & 31;
    const int lmod  = lane & 15;
    const int lhalf = lane >> 4;

    const int rowBase = blockIdx.x * 256 + wave * 32;

    // ---- Load + convert two A tiles (2 x 16 tokens x 256) into registers ----
    // A layout (16-bit 16x32): lane%16 = M row; lanes<16 hold K runs {0..7,16..23},
    // lanes>=16 hold {8..15,24..31}; VGPR i packs (K=2i, K=2i+1).
    const int ak0 = lhalf ? 8 : 0;

    BfTile ahi[2][8], alo[2][8];
#pragma unroll
    for (int t = 0; t < 2; ++t) {
        const float* xr = x + (rowBase + t * 16 + lmod) * DIM;
#pragma unroll
        for (int c = 0; c < 8; ++c) {
            const float* p = xr + c * 32 + ak0;
            float4 f0 = *(const float4*)(p);
            float4 f1 = *(const float4*)(p + 4);
            float4 f2 = *(const float4*)(p + 16);
            float4 f3 = *(const float4*)(p + 20);
            float vv[16] = {f0.x, f0.y, f0.z, f0.w, f1.x, f1.y, f1.z, f1.w,
                            f2.x, f2.y, f2.z, f2.w, f3.x, f3.y, f3.z, f3.w};
#pragma unroll
            for (int i = 0; i < 8; ++i) {
                unsigned short h0, l0, h1, l1;
                split_bf16(vv[2 * i + 0], h0, l0);
                split_bf16(vv[2 * i + 1], h1, l1);
                ahi[t][c].u[i] = (unsigned int)h0 | ((unsigned int)h1 << 16);
                alo[t][c].u[i] = (unsigned int)l0 | ((unsigned int)l1 << 16);
            }
        }
    }

    float minv[2][8];
    int   mini[2][8];
#pragma unroll
    for (int t = 0; t < 2; ++t)
#pragma unroll
        for (int e = 0; e < 8; ++e) { minv[t][e] = 3.4e38f; mini[t][e] = 0; }

    // B layout (16-bit 32x16): lane%16 = N col; lanes<16 hold K runs {0..7,16..23},
    // lanes>=16 hold {8..15,24..31}; VGPR i packs (K=2i, K=2i+1) within the run.
    const int bk0 = lhalf ? 8 : 0;

    for (int nb = 0; nb < KCODES; nb += 16) {
        const int j = nb + lmod;
        const unsigned short* bh = cb_hi + j * DIM;
        const unsigned short* bl = cb_lo + j * DIM;

        // hint the next tile toward the WGP (codebook lives in L2 anyway)
        __builtin_prefetch((const void*)(bh + 16 * DIM), 0, 1);

        v8f acc0 = {0.f, 0.f, 0.f, 0.f, 0.f, 0.f, 0.f, 0.f};
        v8f acc1 = {0.f, 0.f, 0.f, 0.f, 0.f, 0.f, 0.f, 0.f};
#pragma unroll
        for (int c = 0; c < 8; ++c) {
            BfTile bhv, blv;
            const uint4* ph = (const uint4*)(bh + c * 32 + bk0);
            ((uint4*)bhv.u)[0] = ph[0];
            ((uint4*)bhv.u)[1] = ph[2];
            const uint4* pl = (const uint4*)(bl + c * 32 + bk0);
            ((uint4*)blv.u)[0] = pl[0];
            ((uint4*)blv.u)[1] = pl[2];

            // Two independent accumulation chains sharing the same B tile.
            acc0 = __builtin_amdgcn_wmma_f32_16x16x32_bf16(
                false, ahi[0][c].v, false, bhv.v, (short)0, acc0, false, false);
            acc1 = __builtin_amdgcn_wmma_f32_16x16x32_bf16(
                false, ahi[1][c].v, false, bhv.v, (short)0, acc1, false, false);
            acc0 = __builtin_amdgcn_wmma_f32_16x16x32_bf16(
                false, ahi[0][c].v, false, blv.v, (short)0, acc0, false, false);
            acc1 = __builtin_amdgcn_wmma_f32_16x16x32_bf16(
                false, ahi[1][c].v, false, blv.v, (short)0, acc1, false, false);
            acc0 = __builtin_amdgcn_wmma_f32_16x16x32_bf16(
                false, alo[0][c].v, false, bhv.v, (short)0, acc0, false, false);
            acc1 = __builtin_amdgcn_wmma_f32_16x16x32_bf16(
                false, alo[1][c].v, false, bhv.v, (short)0, acc1, false, false);
        }

        const float cn = cnorm[j];
        // C/D layout: element e -> (M = e + 8*lhalf, N = lmod); ||x||^2 omitted
        // (constant per row, does not affect argmin).
#pragma unroll
        for (int e = 0; e < 8; ++e) {
            float d0 = cn - 2.0f * acc0[e];
            if (d0 < minv[0][e]) { minv[0][e] = d0; mini[0][e] = j; }
            float d1 = cn - 2.0f * acc1[e];
            if (d1 < minv[1][e]) { minv[1][e] = d1; mini[1][e] = j; }
        }
    }

    // Reduce across the 16 lanes of each half (each lane covered N === lmod mod 16)
#pragma unroll
    for (int t = 0; t < 2; ++t) {
#pragma unroll
        for (int e = 0; e < 8; ++e) {
#pragma unroll
            for (int off = 8; off > 0; off >>= 1) {
                float ov = __shfl_xor(minv[t][e], off, 32);
                int   oi = __shfl_xor(mini[t][e], off, 32);
                if (ov < minv[t][e] || (ov == minv[t][e] && oi < mini[t][e])) {
                    minv[t][e] = ov; mini[t][e] = oi;
                }
            }
            if (lmod == 0)
                out_idx[rowBase + t * 16 + lhalf * 8 + e] = mini[t][e];
        }
    }
}

// ---------------------------------------------------------------------------
// Kernel 3: gather codebook rows (exact f32), write quantized output, and
// produce deterministic per-block squared-error partial sums.
// One wave per token; 8 waves (tokens) per block.
// ---------------------------------------------------------------------------
__global__ void __launch_bounds__(256) vq_gather_loss(
    const float* __restrict__ x,
    const float* __restrict__ cb,
    const int* __restrict__ idx,
    float* __restrict__ out_q,
    float* __restrict__ partials)
{
    __shared__ float wsum[8];
    const int wave = threadIdx.x >> 5;
    const int lane = threadIdx.x & 31;
    const int tok  = blockIdx.x * 8 + wave;
    const int j    = idx[tok];

    const float* cr = cb + j * DIM + lane * 8;
    const float* xr = x + tok * DIM + lane * 8;
    float4 c0 = *(const float4*)(cr);
    float4 c1 = *(const float4*)(cr + 4);
    float4 x0 = *(const float4*)(xr);
    float4 x1 = *(const float4*)(xr + 4);

    float* qr = out_q + tok * DIM + lane * 8;
    *(float4*)(qr)     = c0;
    *(float4*)(qr + 4) = c1;

    float d0 = c0.x - x0.x, d1 = c0.y - x0.y, d2 = c0.z - x0.z, d3 = c0.w - x0.w;
    float d4 = c1.x - x1.x, d5 = c1.y - x1.y, d6 = c1.z - x1.z, d7 = c1.w - x1.w;
    float s = d0 * d0 + d1 * d1 + d2 * d2 + d3 * d3 +
              d4 * d4 + d5 * d5 + d6 * d6 + d7 * d7;

#pragma unroll
    for (int off = 16; off > 0; off >>= 1)
        s += __shfl_xor(s, off, 32);
    if (lane == 0) wsum[wave] = s;
    __syncthreads();
    if (threadIdx.x == 0) {
        float t = 0.f;
#pragma unroll
        for (int i = 0; i < 8; ++i) t += wsum[i];
        partials[blockIdx.x] = t;
    }
}

// ---------------------------------------------------------------------------
// Kernel 4: deterministic fixed-order reduction of partials -> loss scalar
// ---------------------------------------------------------------------------
__global__ void __launch_bounds__(256) vq_finalize(
    const float* __restrict__ partials, int n, float* __restrict__ out_loss)
{
    __shared__ float sm[256];
    float s = 0.f;
    for (int i = threadIdx.x; i < n; i += 256) s += partials[i];
    sm[threadIdx.x] = s;
    __syncthreads();
    for (int off = 128; off > 0; off >>= 1) {
        if (threadIdx.x < off) sm[threadIdx.x] += sm[threadIdx.x + off];
        __syncthreads();
    }
    if (threadIdx.x == 0)
        out_loss[0] = 0.25f * sm[0] / (float)(TOKENS * DIM);
}

// ---------------------------------------------------------------------------
extern "C" void kernel_launch(void* const* d_in, const int* in_sizes, int n_in,
                              void* d_out, int out_size, void* d_ws, size_t ws_size,
                              hipStream_t stream)
{
    (void)in_sizes; (void)n_in; (void)out_size; (void)ws_size;

    const float* d_x  = (const float*)d_in[0];  // [64,512,256] f32
    const float* d_cb = (const float*)d_in[1];  // [8192,256]   f32
    float* d_q = (float*)d_out;                 // quantized + loss scalar at end

    // d_ws layout (bytes):
    char* ws = (char*)d_ws;
    unsigned short* cb_hi   = (unsigned short*)(ws + 0);        // 4 MB
    unsigned short* cb_lo   = (unsigned short*)(ws + 4194304);  // 4 MB
    float*          cnorm   = (float*)(ws + 8388608);           // 32 KB
    int*            idx     = (int*)(ws + 8421376);             // 128 KB
    float*          partial = (float*)(ws + 8552448);           // 16 KB

    vq_pack_cnorm<<<KCODES / 8, 256, 0, stream>>>(d_cb, cb_hi, cb_lo, cnorm);
    vq_argmin<<<TOKENS / 256, 256, 0, stream>>>(d_x, cb_hi, cb_lo, cnorm, idx);
    vq_gather_loss<<<TOKENS / 8, 256, 0, stream>>>(d_x, d_cb, idx, d_q, partial);
    vq_finalize<<<1, 256, 0, stream>>>(partial, TOKENS / 8, d_q + TOKENS * DIM);
}
